// RvNN_66460323938888
// MI455X (gfx1250) — matrix operand order repaired and checked by reading, not compiled, not run on hardware
//
#include <hip/hip_runtime.h>

typedef __attribute__((ext_vector_type(16))) _Float16 v16h;
typedef __attribute__((ext_vector_type(8)))  float    v8f;

#define WAVES_PER_BLOCK 2
#define XSTRIDE 34              // 16x32 input tile, +2 halfs pad (bank spread)
#define HSTRIDE 130             // 16x128 hidden tile, +2 pad
#define CSTRIDE 514             // 16x512 concat tile, +2 pad
#define LDS_HALFS_PER_WAVE (16 * XSTRIDE + 16 * HSTRIDE + 16 * CSTRIDE)

// packed-weight fragment table offsets (each frag = 512 halfs = 32 lanes x 16)
#define F_WC1 0     // K=24->32 (1 kc) x N=128 (8 nt)
#define F_WC2 8     // K=128 (4 kc) x N=256 (16 nt)
#define F_WP1 72    // K=12->32 (1 kc) x N=128 (8 nt)
#define F_WP2 80    // K=128 (4 kc) x N=256 (16 nt)
#define F_WL1 144   // K=512 (16 kc) x N=128 (8 nt)
#define F_WL2 272   // K=128 (4 kc) x N=32 (2 nt)
#define N_FRAGS 280

// ---------------------------------------------------------------- helpers

__device__ __forceinline__ float san(float x) {
  unsigned u = __float_as_uint(x);
  if ((u & 0x7f800000u) == 0x7f800000u)
    return (u & 0x007fffffu) ? 0.0f : 9999.0f;   // NaN -> 0, +/-Inf -> 9999
  return x;
}

// tanh(x) = 1 - 2/(e^{2x}+1) on the transcendental unit:
// one v_exp_f32 (2^y with the log2(e)*2 scale folded into a v_mul) + one v_rcp_f32.
// Saturates correctly: x>>0 -> exp=inf -> rcp=0 -> 1 ; x<<0 -> exp=0 -> rcp(1)=1 -> -1.
__device__ __forceinline__ float fast_tanh(float x) {
  const float e = __builtin_amdgcn_exp2f(x * 2.885390081777927f);  // e^(2x)
  return 1.0f - 2.0f * __builtin_amdgcn_rcpf(e + 1.0f);
}

// A-matrix 16x32 f16 fragment from LDS (row-major [16][stride] halfs).
// ISA layout: lane l holds row m=l&15; element e -> k = k0 + (e>>3)*16 + (l>>4)*8 + (e&7)
__device__ __forceinline__ v16h load_a_frag(const _Float16* buf, int stride, int k0) {
  const int lane = threadIdx.x & 31;
  const int m  = lane & 15;
  const int hs = (lane >> 4) << 3;
  const _Float16* row = buf + m * stride + k0 + hs;
  v16h a;
#pragma unroll
  for (int p = 0; p < 8; ++p) {
    const int k = ((p >> 2) << 4) + ((p & 3) << 1);  // pairs are k-contiguous
    a[2 * p]     = row[k];
    a[2 * p + 1] = row[k + 1];
  }
  return a;
}

// B-matrix 32x16 f16 fragment from the prepacked table: 16 contiguous halfs/lane.
__device__ __forceinline__ v16h load_b_frag(const _Float16* pack, int frag) {
  const int lane = threadIdx.x & 31;
  const _Float16* p = pack + ((size_t)frag << 9) + (lane << 4);
  return *(const v16h*)p;   // 32B aligned -> 2x global_load_b128
}

__device__ __forceinline__ v8f wmma_f16(v16h a, v16h b, v8f c) {
  return __builtin_amdgcn_wmma_f32_16x16x32_f16(false, a, false, b, (short)0, c,
                                                false, false);
}

// D layout (16x16 f32): lane holds n = n0 + (l&15); VGPR v -> m = v + 8*(l>>4).
__device__ __forceinline__ void store_act_to_lds(_Float16* buf, int stride, int n0,
                                                 v8f c, float bias, bool use_tanh,
                                                 float alpha) {
  const int lane = threadIdx.x & 31;
  const int n  = n0 + (lane & 15);
  const int mb = (lane >> 4) << 3;
#pragma unroll
  for (int vv = 0; vv < 8; ++vv) {
    float x = c[vv] + bias;
    x = use_tanh ? fast_tanh(x) : (x >= 0.0f ? x : alpha * x);
    buf[(mb + vv) * stride + n] = (_Float16)x;
  }
}

// children pair-gather: row r=(b,j): k<12 -> child[b, 2j+1, k], 12<=k<24 -> child[b, 2j, k-12]
__device__ __forceinline__ void gather_child(_Float16* xbuf, const float* src,
                                             int row0, int njshift, int rl) {
  const int lane = threadIdx.x & 31;
  const int k = lane;
  const int nj = 1 << njshift;
#pragma unroll
  for (int m = 0; m < 16; ++m) {
    const int r = row0 + m;
    const int b = r >> njshift;
    const int j = r & (nj - 1);
    float v = 0.0f;
    if (k < 24) {
      const int kk   = (k < 12) ? k : (k - 12);
      const int crow = (b << (njshift + 1)) + 2 * j + ((k < 12) ? 1 : 0);
      v = san(src[(size_t)crow * rl + kk]);
    }
    xbuf[m * XSTRIDE + k] = (_Float16)v;
  }
}

__device__ __forceinline__ void gather_parent(_Float16* xbuf, const float* src,
                                              int row0, int njshift) {
  const int lane = threadIdx.x & 31;
  const int k = lane;
#pragma unroll
  for (int m = 0; m < 16; ++m) {
    const int r = row0 + m;
    float v = 0.0f;
    if (k < 12) v = san(src[(size_t)r * 12 + k]);   // r == b*nj + j
    xbuf[m * XSTRIDE + k] = (_Float16)v;
  }
}

// enc(x) = tanh(prelu(x@W1+b1, a)@W2+b2); x is 16x32 in xbuf; result 16x256 -> cbuf[cofs..]
__device__ __forceinline__ void enc_path(const _Float16* xbuf, _Float16* hbuf,
                                         _Float16* cbuf, int cofs,
                                         const _Float16* wpack, int f_w1,
                                         const float* b1, float alpha, int f_w2,
                                         const float* b2) {
  const int lane = threadIdx.x & 31;
  v16h a0 = load_a_frag(xbuf, XSTRIDE, 0);
#pragma unroll
  for (int nt = 0; nt < 8; ++nt) {
    v8f c = {};
    c = wmma_f16(a0, load_b_frag(wpack, f_w1 + nt), c);
    store_act_to_lds(hbuf, HSTRIDE, nt * 16, c, b1[nt * 16 + (lane & 15)], false,
                     alpha);
  }
  __syncthreads();
  v16h ha[4];
#pragma unroll
  for (int kc = 0; kc < 4; ++kc) ha[kc] = load_a_frag(hbuf, HSTRIDE, kc * 32);
  for (int nt = 0; nt < 16; ++nt) {
    v8f c = {};
#pragma unroll
    for (int kc = 0; kc < 4; ++kc)
      c = wmma_f16(ha[kc], load_b_frag(wpack, f_w2 + kc * 16 + nt), c);
    store_act_to_lds(cbuf, CSTRIDE, cofs + nt * 16, c, b2[nt * 16 + (lane & 15)],
                     true, 0.0f);
  }
}

// ---------------------------------------------------------------- kernels

__global__ void pack_weights_kernel(const float* __restrict__ Wc1,
                                    const float* __restrict__ Wc2,
                                    const float* __restrict__ Wp1,
                                    const float* __restrict__ Wp2,
                                    const float* __restrict__ Wl1,
                                    const float* __restrict__ Wl2,
                                    _Float16* __restrict__ pack) {
  const int t = blockIdx.x * blockDim.x + threadIdx.x;
  if (t >= N_FRAGS * 32) return;
  const int frag = t >> 5;
  const int lane = t & 31;
  const float* W; int K, N, kc, nt;
  if (frag < F_WC2)      { W = Wc1; K = 24;  N = 128; kc = 0;      nt = frag; }
  else if (frag < F_WP1) { int f = frag - F_WC2; W = Wc2; K = 128; N = 256; kc = f >> 4; nt = f & 15; }
  else if (frag < F_WP2) { W = Wp1; K = 12;  N = 128; kc = 0;      nt = frag - F_WP1; }
  else if (frag < F_WL1) { int f = frag - F_WP2; W = Wp2; K = 128; N = 256; kc = f >> 4; nt = f & 15; }
  else if (frag < F_WL2) { int f = frag - F_WL1; W = Wl1; K = 512; N = 128; kc = f >> 3; nt = f & 7; }
  else                   { int f = frag - F_WL2; W = Wl2; K = 128; N = 32;  kc = f >> 1; nt = f & 1; }
  // B layout: lane = n + 16*h, element e -> k = kc*32 + (e>>3)*16 + 8*h + (e&7)
  const int n     = nt * 16 + (lane & 15);
  const int kbase = kc * 32 + ((lane >> 4) << 3);
  _Float16* dst = pack + ((size_t)frag << 9) + (lane << 4);
#pragma unroll
  for (int e = 0; e < 16; ++e) {
    const int k = kbase + ((e >> 3) << 4) + (e & 7);
    dst[e] = (_Float16)((k < K) ? W[(size_t)k * N + n] : 0.0f);
  }
}

__global__ __launch_bounds__(WAVES_PER_BLOCK * 32)
void rvnn_stage_kernel(const float* __restrict__ child, int child_rl, int njshift,
                       const float* __restrict__ parent,
                       const _Float16* __restrict__ wpack,
                       const float* __restrict__ bc1, const float* __restrict__ ac,
                       const float* __restrict__ bc2, const float* __restrict__ bp1,
                       const float* __restrict__ ap, const float* __restrict__ bp2,
                       const float* __restrict__ bl1, const float* __restrict__ bl2,
                       const float* __restrict__ al, float* __restrict__ out,
                       int ntiles) {
  __shared__ _Float16 smem[WAVES_PER_BLOCK * LDS_HALFS_PER_WAVE];
  const int wave = threadIdx.x >> 5;
  const int lane = threadIdx.x & 31;
  const int tile = blockIdx.x * WAVES_PER_BLOCK + wave;
  if (tile >= ntiles) return;
  _Float16* xbuf = smem + wave * LDS_HALFS_PER_WAVE;
  _Float16* hbuf = xbuf + 16 * XSTRIDE;
  _Float16* cbuf = hbuf + 16 * HSTRIDE;
  const int row0 = tile * 16;
  const float a_c = ac[0], a_p = ap[0], a_l = al[0];

  // children path -> cbuf[:, 0:256]
  gather_child(xbuf, child, row0, njshift, child_rl);
  __syncthreads();
  enc_path(xbuf, hbuf, cbuf, 0, wpack, F_WC1, bc1, a_c, F_WC2, bc2);
  __syncthreads();
  // parents path -> cbuf[:, 256:512]
  gather_parent(xbuf, parent, row0, njshift);
  __syncthreads();
  enc_path(xbuf, hbuf, cbuf, 256, wpack, F_WP1, bp1, a_p, F_WP2, bp2);
  __syncthreads();

  // h = tanh(concat @ Wl1 + bl1): K=512, N=128. K-outer, 8 live accumulators.
  v8f acc[8];
#pragma unroll
  for (int nt = 0; nt < 8; ++nt) { v8f z = {}; acc[nt] = z; }
  for (int kc = 0; kc < 16; ++kc) {
    v16h a = load_a_frag(cbuf, CSTRIDE, kc * 32);
#pragma unroll
    for (int nt = 0; nt < 8; ++nt)
      acc[nt] = wmma_f16(a, load_b_frag(wpack, F_WL1 + kc * 8 + nt), acc[nt]);
  }
#pragma unroll
  for (int nt = 0; nt < 8; ++nt)
    store_act_to_lds(hbuf, HSTRIDE, nt * 16, acc[nt], bl1[nt * 16 + (lane & 15)],
                     true, 0.0f);
  __syncthreads();

  // out = prelu(h @ Wl2 + bl2, al): K=128, N=32, f32 stores to global.
  v16h ha[4];
#pragma unroll
  for (int kc = 0; kc < 4; ++kc) ha[kc] = load_a_frag(hbuf, HSTRIDE, kc * 32);
#pragma unroll
  for (int nt = 0; nt < 2; ++nt) {
    v8f c = {};
#pragma unroll
    for (int kc = 0; kc < 4; ++kc)
      c = wmma_f16(ha[kc], load_b_frag(wpack, F_WL2 + kc * 2 + nt), c);
    const float bias = bl2[nt * 16 + (lane & 15)];
    const int n  = nt * 16 + (lane & 15);
    const int mb = (lane >> 4) << 3;
#pragma unroll
    for (int vv = 0; vv < 8; ++vv) {
      float x = c[vv] + bias;
      x = (x >= 0.0f) ? x : a_l * x;
      out[(size_t)(row0 + mb + vv) * 32 + n] = x;
    }
  }
}

// ---------------------------------------------------------------- launch

extern "C" void kernel_launch(void* const* d_in, const int* in_sizes, int n_in,
                              void* d_out, int out_size, void* d_ws, size_t ws_size,
                              hipStream_t stream) {
  const float* level1 = (const float*)d_in[0];
  const float* level2 = (const float*)d_in[1];
  const float* level3 = (const float*)d_in[2];
  const float* Wc1 = (const float*)d_in[3];
  const float* bc1 = (const float*)d_in[4];
  const float* ac  = (const float*)d_in[5];
  const float* Wc2 = (const float*)d_in[6];
  const float* bc2 = (const float*)d_in[7];
  const float* Wp1 = (const float*)d_in[8];
  const float* bp1 = (const float*)d_in[9];
  const float* ap  = (const float*)d_in[10];
  const float* Wp2 = (const float*)d_in[11];
  const float* bp2 = (const float*)d_in[12];
  const float* Wl1 = (const float*)d_in[13];
  const float* bl1 = (const float*)d_in[14];
  const float* Wl2 = (const float*)d_in[15];
  const float* bl2 = (const float*)d_in[16];
  const float* al  = (const float*)d_in[17];

  const int B = in_sizes[0] / 24;           // level1 is (B, 2, 12)
  _Float16* wpack = (_Float16*)d_ws;        // 280 frags * 1 KiB = 280 KiB
  float* out3 = (float*)((char*)d_ws + (512 * 1024));  // (B,4,32) f32 = 16 MiB

  pack_weights_kernel<<<(N_FRAGS * 32 + 255) / 256, 256, 0, stream>>>(
      Wc1, Wc2, Wp1, Wp2, Wl1, Wl2, wpack);

  const int tiles1 = (B * 4) / 16;          // stage 1: level3+level2 -> out3
  rvnn_stage_kernel<<<(tiles1 + WAVES_PER_BLOCK - 1) / WAVES_PER_BLOCK,
                      WAVES_PER_BLOCK * 32, 0, stream>>>(
      level3, 12, 2, level2, wpack, bc1, ac, bc2, bp1, ap, bp2, bl1, bl2, al,
      out3, tiles1);

  const int tiles2 = (B * 2) / 16;          // stage 2: out3+level1 -> d_out
  rvnn_stage_kernel<<<(tiles2 + WAVES_PER_BLOCK - 1) / WAVES_PER_BLOCK,
                      WAVES_PER_BLOCK * 32, 0, stream>>>(
      out3, 32, 1, level1, wpack, bc1, ac, bc2, bp1, ap, bp2, bl1, bl2, al,
      (float*)d_out, tiles2);
}
